// FlowMatching_9268539424829
// MI455X (gfx1250) — matrix operand order repaired and checked by reading, not compiled, not run on hardware
//
#include <hip/hip_runtime.h>
#include <hip/hip_bf16.h>

typedef __attribute__((ext_vector_type(16))) _Float16 v16h;
typedef __attribute__((ext_vector_type(8)))  _Float16 v8h;
typedef __attribute__((ext_vector_type(8)))  float    v8f;

#define NNODES 3072
#define NBATCH 32
#define LSEQ   96
#define HIDC   128
#define LATC   64
#define NLAY   3

enum { ACT_NONE = 0, ACT_RELU = 1, ACT_SILU = 2 };

__device__ __forceinline__ float silu_f(float v) {
    // v * sigmoid(v) with HW reciprocal (v_rcp_f32) instead of precise division
    return v * __builtin_amdgcn_rcpf(1.0f + __expf(-v));
}

// ---------------------------------------------------------------------------
// Generic fp32-in / f16-compute / fp32-out GEMM: C = act(A@B + bias) [+ res]
// 4 waves per block sharing one M-tile and 4 consecutive N-tiles.
// B is cooperatively staged in LDS in WMMA-fragment-major f16 layout, so the
// B operand of each v_wmma is one aligned v16h LDS read (2x ds_load_b128).
// Requires: M%16==0, K%32==0, (N/16)%4==0.
// ---------------------------------------------------------------------------
template <int ACT, bool RES>
__global__ __launch_bounds__(128) void gemm_wmma_kernel(
    const float* __restrict__ A, const float* __restrict__ Bw,
    const float* __restrict__ bias, const float* __restrict__ residual,
    float* __restrict__ C, int M, int N, int K) {
    extern __shared__ char gsm[];
    _Float16* sB = (_Float16*)gsm;   // (K/32) x 4 ntiles x 32 lanes x 16 halves

    const int lane   = threadIdx.x & 31;
    const int wave   = threadIdx.x >> 5;
    const int ntn    = N >> 4;
    const int ntBase = (blockIdx.x * 4) % ntn;
    const int mt     = (blockIdx.x * 4) / ntn;

    // ---- stage B fragments: source-(k,n)-major for coalesced global reads ----
    for (int idx = threadIdx.x; idx < K * 64; idx += 128) {
        const int k  = idx >> 6, nl = idx & 63;
        const int kc = k >> 5, kin = k & 31;
        const int l2 = ((kin >> 3) & 1) * 16 + (nl & 15);
        const int t  = (kin & 7) + ((kin >> 4) << 3);
        const int ntl = nl >> 4;
        sB[((kc * 4 + ntl) * 32 + l2) * 16 + t] =
            (_Float16)Bw[(size_t)k * N + ntBase * 16 + nl];
    }
    __syncthreads();

    const int row_a = lane & 15;            // A row within M-tile
    const int kaoff = (lane < 16) ? 0 : 8;  // A k sub-offset

    v8f acc = {};
    const float* Arow = A + (size_t)(mt * 16 + row_a) * K;
    for (int kc = 0; kc < (K >> 5); ++kc) {
        const int k0 = kc * 32;
        __builtin_prefetch(Arow + k0 + 32, 0, 1);
        const float4 a0 = *(const float4*)(Arow + k0 + kaoff);
        const float4 a1 = *(const float4*)(Arow + k0 + kaoff + 4);
        const float4 a2 = *(const float4*)(Arow + k0 + kaoff + 16);
        const float4 a3 = *(const float4*)(Arow + k0 + kaoff + 20);
        v16h af;
        af[0]  = (_Float16)a0.x; af[1]  = (_Float16)a0.y;
        af[2]  = (_Float16)a0.z; af[3]  = (_Float16)a0.w;
        af[4]  = (_Float16)a1.x; af[5]  = (_Float16)a1.y;
        af[6]  = (_Float16)a1.z; af[7]  = (_Float16)a1.w;
        af[8]  = (_Float16)a2.x; af[9]  = (_Float16)a2.y;
        af[10] = (_Float16)a2.z; af[11] = (_Float16)a2.w;
        af[12] = (_Float16)a3.x; af[13] = (_Float16)a3.y;
        af[14] = (_Float16)a3.z; af[15] = (_Float16)a3.w;
        const v16h bf = *(const v16h*)(sB + ((kc * 4 + wave) * 32 + lane) * 16);
        acc = __builtin_amdgcn_wmma_f32_16x16x32_f16(false, af, false, bf,
                                                     (short)0, acc, false, false);
    }
    const int crb  = (lane < 16) ? 0 : 8;
    const int ncol = (ntBase + wave) * 16 + (lane & 15);
#pragma unroll
    for (int r = 0; r < 8; ++r) {
        const int m = mt * 16 + crb + r;
        float v = acc[r];
        if (bias) v += bias[ncol];
        if (ACT == ACT_RELU) v = fmaxf(v, 0.0f);
        if (ACT == ACT_SILU) v = silu_f(v);
        if (RES) v += residual[(size_t)m * N + ncol];
        C[(size_t)m * N + ncol] = v;
    }
}

static void gemm(const float* A, const float* B, const float* bias,
                 const float* res, float* C, int M, int N, int K, int act,
                 bool resid, hipStream_t s) {
    const int tiles  = (M / 16) * (N / 16);
    const int blocks = tiles / 4;              // exact: (N/16)%4==0, M%16==0
    const size_t smem = (size_t)K * 64 * sizeof(_Float16);
    if (resid)
        gemm_wmma_kernel<ACT_NONE, true><<<blocks, 128, smem, s>>>(A, B, bias, res, C, M, N, K);
    else if (act == ACT_RELU)
        gemm_wmma_kernel<ACT_RELU, false><<<blocks, 128, smem, s>>>(A, B, bias, res, C, M, N, K);
    else if (act == ACT_SILU)
        gemm_wmma_kernel<ACT_SILU, false><<<blocks, 128, smem, s>>>(A, B, bias, res, C, M, N, K);
    else
        gemm_wmma_kernel<ACT_NONE, false><<<blocks, 128, smem, s>>>(A, B, bias, res, C, M, N, K);
}

// ---------------------------------------------------------------------------
// Fused edge kernel. One workgroup = (block b, i-tile of 16 rows).
// LDS-resident: Q (96x128 f32), P (16x128 f32, be1 folded), We2 in
// fragment-swizzled f16 layout, staged m1 tile (256 edges x 128 f16).
// WMMA over 16x16 tiles; coef/agg/dX reduced with LDS atomics. Per-edge data
// never touches HBM.
// ---------------------------------------------------------------------------
constexpr size_t EDGE_SMEM =
    (size_t)(96 * 128 + 16 * 128 + 2 * 128 + 128 + 128 + 128 + 16 * 128 +
             96 * 4 + 16 * 4 + 256 + 16 * 4) * 4   /* floats  */
    + 128 * 4                                       /* ints    */
    + (size_t)(128 * 128 + 256 * 128) * 2;          /* halves  */

__global__ __launch_bounds__(256) void edge_layer_kernel(
    const float* __restrict__ Pg, const float* __restrict__ Qg,
    const float* __restrict__ We1L, const float* __restrict__ We2L,
    const float* __restrict__ be2L, const float* __restrict__ WxL,
    const float* __restrict__ edge_table, const int* __restrict__ chain,
    const float* __restrict__ Xin, float* __restrict__ Xout,
    float* __restrict__ agg) {
    extern __shared__ char smem[];
    float* sQ    = (float*)smem;        // 96*128
    float* sP    = sQ + 96 * 128;       // 16*128
    float* sEd   = sP + 16 * 128;       // 2*128   edge_table @ We1[257:289]
    float* sWc   = sEd + 2 * 128;       // 128     We1 row 256 (d2 row)
    float* sBe2  = sWc + 128;           // 128
    float* sWx   = sBe2 + 128;          // 128
    float* sAgg  = sWx + 128;           // 16*128
    float* sXj   = sAgg + 16 * 128;     // 96*4
    float* sXi   = sXj + 96 * 4;        // 16*4
    float* sCoef = sXi + 16 * 4;        // 256
    float* sDX   = sCoef + 256;         // 16*4
    int* sCidJ   = (int*)(sDX + 16 * 4);          // 96
    int* sCidI   = sCidJ + 96;                    // 16 (pad region to 128 ints)
    _Float16* sW2f = (_Float16*)(sCidJ + 128);    // 4 kc x 8 nt x 32 lane x 16
    _Float16* sM1  = sW2f + 128 * 128;            // 256 edges x 128

    const int b        = blockIdx.x;
    const int it       = blockIdx.y;
    const int nodeBase = b * LSEQ;
    const int i0       = it * 16;
    const int tid      = threadIdx.x;

    // ---- cooperative staging ----
    for (int idx = tid; idx < LSEQ * HIDC; idx += 256)
        sQ[idx] = Qg[(size_t)(nodeBase + (idx >> 7)) * HIDC + (idx & 127)];
    for (int idx = tid; idx < 16 * HIDC; idx += 256)
        sP[idx] = Pg[(size_t)(nodeBase + i0 + (idx >> 7)) * HIDC + (idx & 127)];
    // We2 -> fragment-swizzled f16 (coalesced global reads, scattered LDS writes)
    for (int idx = tid; idx < HIDC * HIDC; idx += 256) {
        const int k  = idx >> 7, n = idx & 127;
        const int kc = k >> 5, kin = k & 31;
        const int l2 = ((kin >> 3) & 1) * 16 + (n & 15);
        const int t  = (kin & 7) + ((kin >> 4) << 3);
        const int ntl = n >> 4;
        sW2f[((kc * 8 + ntl) * 32 + l2) * 16 + t] = (_Float16)We2L[idx];
    }
    {   // Ed[t][c] = sum_e edge_table[t][e] * We1[257+e][c]   (256 = 2*128 exact)
        const int t = tid >> 7, c = tid & 127;
        float s = 0.0f;
        for (int e = 0; e < 32; ++e)
            s += edge_table[t * 32 + e] * We1L[(size_t)(257 + e) * HIDC + c];
        sEd[tid] = s;
    }
    if (tid < HIDC) {
        sWc[tid]  = We1L[(size_t)256 * HIDC + tid];
        sBe2[tid] = be2L[tid];
        sWx[tid]  = WxL[tid];
    }
    if (tid < LSEQ) {
        sXj[tid * 4 + 0] = Xin[(nodeBase + tid) * 3 + 0];
        sXj[tid * 4 + 1] = Xin[(nodeBase + tid) * 3 + 1];
        sXj[tid * 4 + 2] = Xin[(nodeBase + tid) * 3 + 2];
        sCidJ[tid]       = chain[nodeBase + tid];
    }
    if (tid < 16) {
        sXi[tid * 4 + 0] = Xin[(nodeBase + i0 + tid) * 3 + 0];
        sXi[tid * 4 + 1] = Xin[(nodeBase + i0 + tid) * 3 + 1];
        sXi[tid * 4 + 2] = Xin[(nodeBase + i0 + tid) * 3 + 2];
        sCidI[tid]       = chain[nodeBase + i0 + tid];
        sDX[tid * 4 + 0] = 0.0f; sDX[tid * 4 + 1] = 0.0f;
        sDX[tid * 4 + 2] = 0.0f; sDX[tid * 4 + 3] = 0.0f;
    }
    for (int idx = tid; idx < 16 * HIDC; idx += 256) sAgg[idx] = 0.0f;
    __syncthreads();

    const int lane  = tid & 31;
    const int nt    = tid >> 5;              // wave id == N-tile (8 x 16 = 128 cols)
    const int ar    = lane & 15;             // A row / B col / C col within tile
    const int kaoff = (lane < 16) ? 0 : 8;
    const int crb   = (lane < 16) ? 0 : 8;

    for (int jt = 0; jt < LSEQ / 16; ++jt) {
        const int j0 = jt * 16;
        sCoef[tid] = 0.0f;
        // ---- stage m1 = silu(P[i] + Q[j] + d2*Wc + Ed[type]) as f16 ----
        {
            const int e  = tid;               // 256 edges: e = ii*16 + jj
            const int ii = e >> 4, jj = e & 15;
            const int j  = j0 + jj;
            const float rx = sXi[ii * 4 + 0] - sXj[j * 4 + 0];
            const float ry = sXi[ii * 4 + 1] - sXj[j * 4 + 1];
            const float rz = sXi[ii * 4 + 2] - sXj[j * 4 + 2];
            const float d2 = rx * rx + ry * ry + rz * rz;
            const float* ed = sEd + ((sCidI[ii] != sCidJ[j]) ? HIDC : 0);
            const float* p  = sP + ii * HIDC;
            const float* q  = sQ + j * HIDC;
            _Float16* dst   = sM1 + e * HIDC;
            for (int c0 = 0; c0 < HIDC; c0 += 8) {
                v8h out;
#pragma unroll
                for (int u = 0; u < 8; ++u) {
                    const int c = c0 + u;
                    const float v = p[c] + q[c] + d2 * sWc[c] + ed[c];
                    out[u] = (_Float16)silu_f(v);
                }
                *(v8h*)(dst + c0) = out;   // ds_store_b128
            }
        }
        __syncthreads();
        // ---- m2 = silu(m1 @ We2 + be2) via WMMA; 16 edge-tiles per wave ----
        for (int et = 0; et < 16; ++et) {
            v8f acc = {};
            const _Float16* am = sM1 + (size_t)(et * 16 + ar) * HIDC;
#pragma unroll
            for (int kc = 0; kc < 4; ++kc) {
                const int k0 = kc * 32;
                v16h af;
#pragma unroll
                for (int t = 0; t < 8; ++t) {
                    af[t]     = am[k0 + kaoff + t];
                    af[t + 8] = am[k0 + kaoff + 16 + t];
                }
                const v16h bf =
                    *(const v16h*)(sW2f + ((kc * 8 + nt) * 32 + lane) * 16);
                acc = __builtin_amdgcn_wmma_f32_16x16x32_f16(
                    false, af, false, bf, (short)0, acc, false, false);
            }
            const int col = nt * 16 + ar;
            const float wx = sWx[col];
            const float b2 = sBe2[col];
            float aggsum = 0.0f;
#pragma unroll
            for (int rr = 0; rr < 8; ++rr) {
                const int jj = crb + rr;   // edge row within tile == j index
                const float v = silu_f(acc[rr] + b2);
                aggsum += v;
                atomicAdd(&sCoef[et * 16 + jj], v * wx);   // coef = m2 @ Wx
            }
            atomicAdd(&sAgg[et * HIDC + col], aggsum);     // agg over this lane's rows
        }
        __syncthreads();
        // ---- X displacement: dX[i] += (X[i]-X[j]) * coef[i,j] ----
        {
            const int e  = tid;
            const int ii = e >> 4, jj = e & 15;
            const int j  = j0 + jj;
            const float cf = sCoef[e];
            atomicAdd(&sDX[ii * 4 + 0], (sXi[ii * 4 + 0] - sXj[j * 4 + 0]) * cf);
            atomicAdd(&sDX[ii * 4 + 1], (sXi[ii * 4 + 1] - sXj[j * 4 + 1]) * cf);
            atomicAdd(&sDX[ii * 4 + 2], (sXi[ii * 4 + 2] - sXj[j * 4 + 2]) * cf);
        }
        __syncthreads();
    }
    // ---- store agg rows and updated X (each node owned by exactly one WG) ----
    for (int idx = tid; idx < 16 * HIDC; idx += 256) {
        const int ii = idx >> 7, c = idx & 127;
        agg[(size_t)(nodeBase + i0 + ii) * HIDC + c] = sAgg[idx];
    }
    if (tid < 16 * 3) {
        const int ii = tid / 3, d = tid % 3;
        Xout[(nodeBase + i0 + ii) * 3 + d] = sXi[ii * 4 + d] + sDX[ii * 4 + d];
    }
}

// ---------------------------------------------------------------------------
// Node prep: t lookup, H_t/X_t interpolation, sinusoidal t-embed, build the
// 320-wide node feature row [H_t | cond | sin | cos].
// ---------------------------------------------------------------------------
__global__ void node_prep_kernel(const float* __restrict__ H0,
                                 const float* __restrict__ X0,
                                 const float* __restrict__ H1,
                                 const float* __restrict__ X1,
                                 const float* __restrict__ cond,
                                 const float* __restrict__ tg,
                                 const unsigned char* __restrict__ mask,
                                 const int* __restrict__ lengths,
                                 float* __restrict__ nf, float* __restrict__ Xt,
                                 int n, int bcount) {
    const int node = blockIdx.x * blockDim.x + threadIdx.x;
    if (node >= n) return;
    int acc = 0, bid = 0;
    for (int bb = 0; bb < bcount; ++bb) {
        if (node >= acc) bid = bb;
        acc += lengths[bb];
    }
    const float t  = tg[bid];
    const float th = mask[node] ? t : 1.0f;   // unmasked rows use H_1/X_1 directly
    float* row = nf + (size_t)node * 320;
    for (int c = 0; c < LATC; ++c)
        row[c] = (1.0f - th) * H0[(size_t)node * LATC + c] + th * H1[(size_t)node * LATC + c];
    for (int c = 0; c < HIDC; ++c)
        row[LATC + c] = cond[(size_t)node * HIDC + c];
    const float nlog = -9.210340371976184f;   // -log(10000)
    for (int k = 0; k < 64; ++k) {
        const float fr  = __expf(nlog * (float)k / 63.0f);
        const float ang = t * fr;
        row[192 + k]      = __sinf(ang);
        row[192 + 64 + k] = __cosf(ang);
    }
    for (int d = 0; d < 3; ++d)
        Xt[node * 3 + d] = (1.0f - th) * X0[node * 3 + d] + th * X1[node * 3 + d];
}

__global__ void concat_kernel(const float* __restrict__ h,
                              const float* __restrict__ agg,
                              float* __restrict__ hh, int n) {
    const int idx = blockIdx.x * blockDim.x + threadIdx.x;
    if (idx >= n * 256) return;
    const int node = idx >> 8, c = idx & 255;
    hh[idx] = (c < HIDC) ? h[(size_t)node * HIDC + c]
                         : agg[(size_t)node * HIDC + (c - HIDC)];
}

__global__ void zero_kernel(float* a, int n) {
    const int i = blockIdx.x * blockDim.x + threadIdx.x;
    if (i < n) a[i] = 0.0f;
}

__global__ void loss_reduce_kernel(const float* __restrict__ vH,
                                   const float* __restrict__ Xf,
                                   const float* __restrict__ H0,
                                   const float* __restrict__ H1,
                                   const float* __restrict__ X0,
                                   const float* __restrict__ X1,
                                   const unsigned char* __restrict__ mask,
                                   const int* __restrict__ shiftp,
                                   float* __restrict__ accum, int n) {
    const int node = blockIdx.x * blockDim.x + threadIdx.x;
    if (node >= n) return;
    const float mf = mask[node] ? 1.0f : 0.0f;
    int sh = shiftp[0] % n;
    if (sh < 0) sh += n;
    int nn = node + sh;
    if (nn >= n) nn -= n;                  // roll(-shift)[i] = src[(i+shift)%n]
    float hp = 0.0f, hn = 0.0f;
    for (int c = 0; c < LATC; ++c) {
        const float pred = mf * vH[(size_t)node * LATC + c];
        const float tp = H1[(size_t)node * LATC + c] - H0[(size_t)node * LATC + c];
        const float tn = H1[(size_t)nn * LATC + c] - H0[(size_t)nn * LATC + c];
        const float d1 = pred - tp, d2 = pred - tn;
        hp += d1 * d1; hn += d2 * d2;
    }
    float xp = 0.0f, xn = 0.0f;
    for (int d = 0; d < 3; ++d) {
        const float pred = mf * Xf[node * 3 + d];
        const float tp = X1[node * 3 + d] - X0[node * 3 + d];
        const float tn = X1[nn * 3 + d] - X0[nn * 3 + d];
        const float d1 = pred - tp, d2 = pred - tn;
        xp += d1 * d1; xn += d2 * d2;
    }
    atomicAdd(&accum[0], hp * mf);
    atomicAdd(&accum[1], xp * mf);
    atomicAdd(&accum[2], hn * mf);
    atomicAdd(&accum[3], xn * mf);
    atomicAdd(&accum[4], mf);
}

__global__ void finalize_kernel(const float* __restrict__ accum,
                                float* __restrict__ out) {
    if (threadIdx.x == 0) {
        const float msum = accum[4] + 1e-8f;
        const float lhp = accum[0] / msum, lxp = accum[1] / msum;
        const float lhn = accum[2] / msum, lxn = accum[3] / msum;
        out[0] = lhp - 0.05f * lhn;
        out[1] = lxp - 0.05f * lxn;
        out[2] = lhp; out[3] = lxp; out[4] = lhn; out[5] = lxn;
    }
}

// ---------------------------------------------------------------------------
extern "C" void kernel_launch(void* const* d_in, const int* in_sizes, int n_in,
                              void* d_out, int out_size, void* d_ws,
                              size_t ws_size, hipStream_t stream) {
    const float* H0   = (const float*)d_in[0];
    const float* X0   = (const float*)d_in[1];
    const float* H1   = (const float*)d_in[2];
    const float* X1   = (const float*)d_in[3];
    const float* cond = (const float*)d_in[4];
    const float* tg   = (const float*)d_in[5];
    const float* Wi1  = (const float*)d_in[6];
    const float* bi1  = (const float*)d_in[7];
    const float* Wi2  = (const float*)d_in[8];
    const float* bi2  = (const float*)d_in[9];
    const float* Wi3  = (const float*)d_in[10];
    const float* bi3  = (const float*)d_in[11];
    const float* edge_table = (const float*)d_in[12];
    const float* We1  = (const float*)d_in[13];
    const float* be1  = (const float*)d_in[14];
    const float* We2  = (const float*)d_in[15];
    const float* be2  = (const float*)d_in[16];
    const float* Wx   = (const float*)d_in[17];
    const float* Wh1  = (const float*)d_in[18];
    const float* bh1  = (const float*)d_in[19];
    const float* Wh2  = (const float*)d_in[20];
    const float* bh2  = (const float*)d_in[21];
    const float* Wout = (const float*)d_in[22];
    const float* bout = (const float*)d_in[23];
    // d_in[24] = edges: unused, structure is dense 96x96 per block
    const int* chain           = (const int*)d_in[25];
    const unsigned char* mask  = (const unsigned char*)d_in[26];
    const int* lengths         = (const int*)d_in[27];
    const int* shiftp          = (const int*)d_in[28];

    float* ws   = (float*)d_ws;
    float* nf   = ws; ws += (size_t)NNODES * 320;
    float* h    = ws; ws += (size_t)NNODES * HIDC;
    float* tmp  = ws; ws += (size_t)NNODES * HIDC;
    float* Pm   = ws; ws += (size_t)NNODES * HIDC;
    float* Qm   = ws; ws += (size_t)NNODES * HIDC;
    float* aggw = ws; ws += (size_t)NNODES * HIDC;
    float* hh   = ws; ws += (size_t)NNODES * 2 * HIDC;
    float* vH   = ws; ws += (size_t)NNODES * LATC;
    float* Xa   = ws; ws += (size_t)NNODES * 3;
    float* Xb   = ws; ws += (size_t)NNODES * 3;
    float* accum = ws; ws += 8;

    (void)hipFuncSetAttribute((const void*)edge_layer_kernel,
                              hipFuncAttributeMaxDynamicSharedMemorySize,
                              (int)EDGE_SMEM);

    zero_kernel<<<1, 32, 0, stream>>>(accum, 8);
    node_prep_kernel<<<(NNODES + 127) / 128, 128, 0, stream>>>(
        H0, X0, H1, X1, cond, tg, mask, lengths, nf, Xa, NNODES, NBATCH);

    // node embedding MLP
    gemm(nf, Wi1, bi1, nullptr, h, NNODES, HIDC, 320, ACT_RELU, false, stream);
    gemm(h, Wi2, bi2, nullptr, tmp, NNODES, HIDC, HIDC, ACT_RELU, false, stream);
    gemm(tmp, Wi3, bi3, nullptr, h, NNODES, HIDC, HIDC, ACT_NONE, false, stream);

    float* Xin = Xa;
    float* Xout = Xb;
    for (int l = 0; l < NLAY; ++l) {
        const float* We1L = We1 + (size_t)l * 289 * HIDC;
        // P = h @ We1[0:128] + be1  (row part);  Q = h @ We1[128:256] (col part)
        gemm(h, We1L, be1 + l * HIDC, nullptr, Pm, NNODES, HIDC, HIDC, ACT_NONE, false, stream);
        gemm(h, We1L + (size_t)128 * HIDC, nullptr, nullptr, Qm, NNODES, HIDC, HIDC, ACT_NONE, false, stream);
        edge_layer_kernel<<<dim3(NBATCH, LSEQ / 16), 256, EDGE_SMEM, stream>>>(
            Pm, Qm, We1L, We2 + (size_t)l * HIDC * HIDC,
            be2 + l * HIDC, Wx + l * HIDC, edge_table, chain, Xin, Xout, aggw);
        concat_kernel<<<(NNODES * 256 + 255) / 256, 256, 0, stream>>>(h, aggw, hh, NNODES);
        gemm(hh, Wh1 + (size_t)l * 2 * HIDC * HIDC, bh1 + l * HIDC, nullptr,
             tmp, NNODES, HIDC, 2 * HIDC, ACT_SILU, false, stream);
        gemm(tmp, Wh2 + (size_t)l * HIDC * HIDC, bh2 + l * HIDC, h, h,
             NNODES, HIDC, HIDC, ACT_NONE, true, stream);
        float* t2 = Xin; Xin = Xout; Xout = t2;
    }

    gemm(h, Wout, bout, nullptr, vH, NNODES, LATC, HIDC, ACT_NONE, false, stream);
    loss_reduce_kernel<<<(NNODES + 255) / 256, 256, 0, stream>>>(
        vH, Xin, H0, H1, X0, X1, mask, shiftp, accum, NNODES);
    finalize_kernel<<<1, 32, 0, stream>>>(accum, (float*)d_out);
}
